// TripleViewNet_13091060318415
// MI455X (gfx1250) — matrix operand order repaired and checked by reading, not compiled, not run on hardware
//
#include <hip/hip_runtime.h>
#include <hip/hip_bf16.h>
#include <hip/hip_fp16.h>

#define NN 100000
#define EE 1600000
#define BBG 256
#define DD 128
#define HH 4
#define CCH 32
#define EN (EE + NN)

typedef __attribute__((ext_vector_type(16))) _Float16 v16h;
typedef __attribute__((ext_vector_type(8)))  _Float16 v8h;
typedef __attribute__((ext_vector_type(8)))  float    v8f;
typedef __attribute__((ext_vector_type(4)))  float    v4f;

// ---------- monotonic float<->uint encoding for atomicMax on floats ----------
__device__ __forceinline__ unsigned ordf(float f) {
    unsigned u = __float_as_uint(f);
    return (u & 0x80000000u) ? ~u : (u | 0x80000000u);
}
__device__ __forceinline__ float unordf(unsigned u) {
    unsigned v = (u & 0x80000000u) ? (u & 0x7fffffffu) : ~u;
    return __uint_as_float(v);
}

// ---------- zero kernels (graph-capture-safe init) ----------
__global__ void k_zero_f(float* p, long n) {
    long i = (long)blockIdx.x * blockDim.x + threadIdx.x;
    if (i < n) p[i] = 0.f;
}
__global__ void k_zero_u(unsigned* p, long n) {
    long i = (long)blockIdx.x * blockDim.x + threadIdx.x;
    if (i < n) p[i] = 0u;
}

// ---------- LayerNorm -> f16, one wave32 per row ----------
__global__ void k_layernorm(const float* __restrict__ x,
                            const float* __restrict__ g,
                            const float* __restrict__ b,
                            _Float16* __restrict__ xnh) {
    int wave = threadIdx.x >> 5, lane = threadIdx.x & 31;
    int row = blockIdx.x * (blockDim.x >> 5) + wave;
    if (row >= NN) return;
    const float* xr = x + (size_t)row * DD;
    float v[4], s = 0.f, s2 = 0.f;
#pragma unroll
    for (int i = 0; i < 4; ++i) { v[i] = xr[lane * 4 + i]; s += v[i]; s2 += v[i] * v[i]; }
#pragma unroll
    for (int m = 16; m >= 1; m >>= 1) { s += __shfl_xor(s, m, 32); s2 += __shfl_xor(s2, m, 32); }
    float mu = s * (1.f / DD);
    float var = s2 * (1.f / DD) - mu * mu;
    float r = rsqrtf(var + 1e-5f);
#pragma unroll
    for (int i = 0; i < 4; ++i) {
        int c = lane * 4 + i;
        xnh[(size_t)row * DD + c] = (_Float16)((v[i] - mu) * r * g[c] + b[c]);
    }
}

// ---------- pack [128,128] f32 weight (k-major) into WMMA B-fragment order ----------
// index: ((ks*8 + nt)*32 + lane)*16 + j ; col=nt*16+(lane&15), h=lane>>4
//   k = ks*32 + 8h + j (j<8), k = ks*32 + 16 + 8h + (j-8) (j>=8)
__global__ void k_pack_w(const float* __restrict__ W, _Float16* __restrict__ Wp) {
    int tid = blockIdx.x * blockDim.x + threadIdx.x;
    if (tid >= DD * DD) return;
    int j = tid & 15, lane = (tid >> 4) & 31, nt = (tid >> 9) & 7, ks = tid >> 12;
    int col = nt * 16 + (lane & 15);
    int kk  = ks * 32 + ((lane >> 4) * 8) + (j & 7) + ((j >> 3) * 16);
    Wp[tid] = (_Float16)W[kk * DD + col];
}

// ---------- WMMA GEMM: [M,128] f16 @ [128,128] f16 (+bias, act) -> f32 / f16 ----------
__global__ __launch_bounds__(128) void k_gemm_wmma(
        const _Float16* __restrict__ A, const _Float16* __restrict__ Wp,
        const float* __restrict__ bias, float* __restrict__ Of,
        _Float16* __restrict__ Oh, int M, int act) {
    int wave = threadIdx.x >> 5, lane = threadIdx.x & 31;
    int rt = blockIdx.x * (blockDim.x >> 5) + wave;
    if (rt * 16 >= M) return;
    int row = rt * 16 + (lane & 15);
    int kb  = (lane >> 4) * 8;
    v8f acc[8] = {};
#pragma unroll
    for (int ks = 0; ks < 4; ++ks) {
        const _Float16* ap = A + (size_t)row * DD + ks * 32 + kb;
        v8h a0 = *(const v8h*)ap;          // K = kb..kb+7
        v8h a1 = *(const v8h*)(ap + 16);   // K = kb+16..kb+23
        v16h a;
#pragma unroll
        for (int j = 0; j < 8; ++j) { a[j] = a0[j]; a[j + 8] = a1[j]; }
#pragma unroll
        for (int nt = 0; nt < 8; ++nt) {
            v16h bfrag = *(const v16h*)(Wp + (((ks * 8 + nt) * 32 + lane) << 4));
            acc[nt] = __builtin_amdgcn_wmma_f32_16x16x32_f16(
                false, a, false, bfrag, (short)0, acc[nt], false, false);
        }
    }
    int n0 = lane & 15;
    int mb = rt * 16 + ((lane >> 4) * 8);
#pragma unroll
    for (int nt = 0; nt < 8; ++nt) {
        int n = nt * 16 + n0;
        float bv = bias ? bias[n] : 0.f;
#pragma unroll
        for (int r = 0; r < 8; ++r) {
            float vv = acc[nt][r] + bv;
            if (act == 1) vv = vv > 0.f ? vv : 0.01f * vv;
            size_t o = (size_t)(mb + r) * DD + n;
            if (Of) Of[o] = vv;
            if (Oh) Oh[o] = (_Float16)vv;
        }
    }
}

// ---------- per-node per-head attention coefficients ----------
__global__ void k_att(const float* __restrict__ xh, const float* __restrict__ as,
                      const float* __restrict__ ad, float* __restrict__ a_src,
                      float* __restrict__ a_dst) {
    int t = blockIdx.x * blockDim.x + threadIdx.x;
    if (t >= NN * HH) return;
    int n = t >> 2, h = t & 3;
    const float* xr = xh + (size_t)n * DD + h * CCH;
    float s1 = 0.f, s2 = 0.f;
#pragma unroll 8
    for (int c = 0; c < CCH; ++c) { float v = xr[c]; s1 += v * as[h * CCH + c]; s2 += v * ad[h * CCH + c]; }
    a_src[t] = s1; a_dst[t] = s2;
}

__device__ __forceinline__ void edge_sd(const int* ei, long idx, int& s, int& d) {
    if (idx < EE) { s = ei[idx]; d = ei[EE + idx]; }
    else { s = d = (int)(idx - EE); }   // self loops
}

// ---------- segment softmax over edges: max, exp+sum, scatter ----------
__global__ void k_edge_max(const int* __restrict__ ei, const float* __restrict__ a_src,
                           const float* __restrict__ a_dst, unsigned* __restrict__ mx) {
    long idx = (long)blockIdx.x * blockDim.x + threadIdx.x;
    if (idx >= EN) return;
    int s, d; edge_sd(ei, idx, s, d);
#pragma unroll
    for (int h = 0; h < HH; ++h) {
        float l = a_src[s * HH + h] + a_dst[d * HH + h];
        l = l > 0.f ? l : 0.2f * l;
        atomicMax(&mx[d * HH + h], ordf(l));
    }
}

__global__ void k_edge_expsum(const int* __restrict__ ei, const float* __restrict__ a_src,
                              const float* __restrict__ a_dst, const unsigned* __restrict__ mx,
                              float* __restrict__ ev, float* __restrict__ sm) {
    long idx = (long)blockIdx.x * blockDim.x + threadIdx.x;
    if (idx >= EN) return;
    int s, d; edge_sd(ei, idx, s, d);
#pragma unroll
    for (int h = 0; h < HH; ++h) {
        float l = a_src[s * HH + h] + a_dst[d * HH + h];
        l = l > 0.f ? l : 0.2f * l;
        float e = __expf(l - unordf(mx[d * HH + h]));
        ev[idx * HH + h] = e;
        atomicAdd(&sm[d * HH + h], e);
    }
}

// one wave32 per edge; lane owns 4 of 128 channels (head = lane>>3)
__global__ void k_edge_scatter(const int* __restrict__ ei, const float* __restrict__ xh,
                               const float* __restrict__ ev, const float* __restrict__ sm,
                               float* __restrict__ acc) {
    int wave = threadIdx.x >> 5, lane = threadIdx.x & 31;
    long idx = (long)blockIdx.x * (blockDim.x >> 5) + wave;
    if (idx >= EN) return;
    int s, d; edge_sd(ei, idx, s, d);
    int h = lane >> 3;
    float sv = sm[d * HH + h];
    float al = sv > 0.f ? ev[idx * HH + h] / sv : 0.f;
    v4f xs = *(const v4f*)(xh + (size_t)s * DD + lane * 4);
    float* ab = acc + (size_t)d * DD + lane * 4;
    atomicAdd(ab + 0, xs[0] * al);
    atomicAdd(ab + 1, xs[1] * al);
    atomicAdd(ab + 2, xs[2] * al);
    atomicAdd(ab + 3, xs[3] * al);
}

__global__ void k_bias_elu(float* __restrict__ acc, const float* __restrict__ bias) {
    long i = (long)blockIdx.x * blockDim.x + threadIdx.x;
    if (i >= (long)NN * DD) return;
    float v = acc[i] + bias[i & (DD - 1)];
    acc[i] = v > 0.f ? v : expm1f(v);
}

// ---------- global attention pooling ----------
__global__ void k_gate(const float* __restrict__ h, const float* __restrict__ gw,
                       const float* __restrict__ gb, float* __restrict__ gate) {
    int n = blockIdx.x * blockDim.x + threadIdx.x;
    if (n >= NN) return;
    const float* hr = h + (size_t)n * DD;
    float s = gb[0];
#pragma unroll 8
    for (int c = 0; c < DD; ++c) s += hr[c] * gw[c];
    gate[n] = s;
}
__global__ void k_gate_max(const float* __restrict__ gate, const int* __restrict__ batch,
                           unsigned* __restrict__ gm) {
    int n = blockIdx.x * blockDim.x + threadIdx.x;
    if (n >= NN) return;
    atomicMax(&gm[batch[n]], ordf(gate[n]));
}
__global__ void k_gate_exp(const float* __restrict__ gate, const int* __restrict__ batch,
                           const unsigned* __restrict__ gm, float* __restrict__ ge,
                           float* __restrict__ gs) {
    int n = blockIdx.x * blockDim.x + threadIdx.x;
    if (n >= NN) return;
    int b = batch[n];
    float e = __expf(gate[n] - unordf(gm[b]));
    ge[n] = e;
    atomicAdd(&gs[b], e);
}
// one wave32 per node
__global__ void k_gate_scatter(const float* __restrict__ h, const int* __restrict__ batch,
                               const float* __restrict__ ge, const float* __restrict__ gs,
                               float* __restrict__ gout) {
    int wave = threadIdx.x >> 5, lane = threadIdx.x & 31;
    long n = (long)blockIdx.x * (blockDim.x >> 5) + wave;
    if (n >= NN) return;
    int b = batch[n];
    float sv = gs[b];
    float a = sv > 0.f ? ge[n] / sv : 0.f;
    v4f hv = *(const v4f*)(h + (size_t)n * DD + lane * 4);
    float* gp = gout + (size_t)b * DD + lane * 4;
    atomicAdd(gp + 0, a * hv[0]);
    atomicAdd(gp + 1, a * hv[1]);
    atomicAdd(gp + 2, a * hv[2]);
    atomicAdd(gp + 3, a * hv[3]);
}

// ---------- tiny classifier head: [256,384]@[384,128] -> leaky -> @[128,1] ----------
__global__ void k_clf(const float* __restrict__ gout, const float* __restrict__ W1,
                      const float* __restrict__ b1, const float* __restrict__ W2,
                      const float* __restrict__ b2, float* __restrict__ out) {
    __shared__ float sh[DD];
    int b = blockIdx.x, j = threadIdx.x;
    float s = b1[j];
    for (int k = 0; k < 3 * DD; ++k) {
        float cv = gout[(size_t)(k >> 7) * (BBG * DD) + (size_t)b * DD + (k & 127)];
        s += cv * W1[k * DD + j];
    }
    s = s > 0.f ? s : 0.01f * s;
    sh[j] = s * W2[j];
    __syncthreads();
    for (int off = 64; off >= 1; off >>= 1) {
        if (j < off) sh[j] += sh[j + off];
        __syncthreads();
    }
    if (j == 0) out[b] = sh[0] + b2[0];
}

extern "C" void kernel_launch(void* const* d_in, const int* in_sizes, int n_in,
                              void* d_out, int out_size, void* d_ws, size_t ws_size,
                              hipStream_t stream) {
    (void)in_sizes; (void)n_in; (void)out_size; (void)ws_size;
    const float* x      = (const float*)d_in[0];
    const int*   batch  = (const int*)d_in[4];
    const float* ln_g   = (const float*)d_in[5];
    const float* ln_b   = (const float*)d_in[6];
    const float* proj_W = (const float*)d_in[7];
    const float* proj_b = (const float*)d_in[8];
    const float* gate_W = (const float*)d_in[21];
    const float* gate_b = (const float*)d_in[22];
    const float* clf_W1 = (const float*)d_in[23];
    const float* clf_b1 = (const float*)d_in[24];
    const float* clf_W2 = (const float*)d_in[25];
    const float* clf_b2 = (const float*)d_in[26];
    float* out = (float*)d_out;

    // ---- workspace layout (256B aligned) ----
    char* w = (char*)d_ws;
    size_t off = 0;
    auto alloc = [&](size_t bytes) -> void* {
        off = (off + 255) & ~(size_t)255;
        void* p = w + off;
        off += bytes;
        return p;
    };
    _Float16* xnh  = (_Float16*)alloc((size_t)NN * DD * 2);
    _Float16* h0h  = (_Float16*)alloc((size_t)NN * DD * 2);
    _Float16* wp   = (_Float16*)alloc((size_t)4 * DD * DD * 2);  // proj + 3 views
    float*    xh   = (float*)alloc((size_t)NN * DD * 4);
    float*    a_src= (float*)alloc((size_t)NN * HH * 4);
    float*    a_dst= (float*)alloc((size_t)NN * HH * 4);
    unsigned* mx   = (unsigned*)alloc((size_t)NN * HH * 4);
    float*    sm   = (float*)alloc((size_t)NN * HH * 4);
    float*    ev   = (float*)alloc((size_t)EN * HH * 4);
    float*    acc  = (float*)alloc((size_t)NN * DD * 4);
    float*    gate = (float*)alloc((size_t)NN * 4);
    float*    gexp = (float*)alloc((size_t)NN * 4);
    unsigned* gm   = (unsigned*)alloc((size_t)BBG * 4);
    float*    gs   = (float*)alloc((size_t)BBG * 4);
    float*    gout = (float*)alloc((size_t)3 * BBG * DD * 4);

    const int TPB = 256;
    const int packGrid = (DD * DD + TPB - 1) / TPB;
    const int gemmGrid = ((NN + 15) / 16 + 3) / 4;     // 4 waves/block
    const int edgeGrid = (EN + TPB - 1) / TPB;
    const int edgeWGrid = (EN + 7) / 8;                // 8 waves/block, wave per edge
    const int nodeGrid = (NN + TPB - 1) / TPB;
    const int nodeWGrid = (NN + 7) / 8;

    // pack weights to WMMA fragment order (f32 -> f16)
    k_pack_w<<<packGrid, TPB, 0, stream>>>(proj_W, wp);
    k_pack_w<<<packGrid, TPB, 0, stream>>>((const float*)d_in[9],  wp + 1 * DD * DD);
    k_pack_w<<<packGrid, TPB, 0, stream>>>((const float*)d_in[13], wp + 2 * DD * DD);
    k_pack_w<<<packGrid, TPB, 0, stream>>>((const float*)d_in[17], wp + 3 * DD * DD);

    // LayerNorm -> f16, then h0 = leaky(xn @ proj_W + b) via WMMA -> f16
    k_layernorm<<<(NN + 3) / 4, 128, 0, stream>>>(x, ln_g, ln_b, xnh);
    k_gemm_wmma<<<gemmGrid, 128, 0, stream>>>(xnh, wp, proj_b, nullptr, h0h, NN, 1);

    k_zero_f<<<(3 * BBG * DD + TPB - 1) / TPB, TPB, 0, stream>>>(gout, 3 * BBG * DD);

    for (int v = 0; v < 3; ++v) {
        const int*   ei  = (const int*)d_in[1 + v];
        const float* asv = (const float*)d_in[10 + 4 * v];
        const float* adv = (const float*)d_in[11 + 4 * v];
        const float* bsv = (const float*)d_in[12 + 4 * v];
        const _Float16* wpv = wp + (size_t)(1 + v) * DD * DD;

        // xh = h0 @ W_view  (WMMA, f32 out)
        k_gemm_wmma<<<gemmGrid, 128, 0, stream>>>(h0h, wpv, nullptr, xh, nullptr, NN, 0);
        k_att<<<(NN * HH + TPB - 1) / TPB, TPB, 0, stream>>>(xh, asv, adv, a_src, a_dst);

        k_zero_u<<<(NN * HH + TPB - 1) / TPB, TPB, 0, stream>>>(mx, (long)NN * HH);
        k_zero_f<<<(NN * HH + TPB - 1) / TPB, TPB, 0, stream>>>(sm, (long)NN * HH);
        k_zero_f<<<((long)NN * DD + TPB - 1) / TPB, TPB, 0, stream>>>(acc, (long)NN * DD);

        k_edge_max<<<edgeGrid, TPB, 0, stream>>>(ei, a_src, a_dst, mx);
        k_edge_expsum<<<edgeGrid, TPB, 0, stream>>>(ei, a_src, a_dst, mx, ev, sm);
        k_edge_scatter<<<edgeWGrid, TPB, 0, stream>>>(ei, xh, ev, sm, acc);
        k_bias_elu<<<((long)NN * DD + TPB - 1) / TPB, TPB, 0, stream>>>(acc, bsv);

        // global attention pooling for this view
        k_gate<<<nodeGrid, TPB, 0, stream>>>(acc, gate_W, gate_b, gate);
        k_zero_u<<<1, TPB, 0, stream>>>(gm, BBG);
        k_zero_f<<<1, TPB, 0, stream>>>(gs, BBG);
        k_gate_max<<<nodeGrid, TPB, 0, stream>>>(gate, batch, gm);
        k_gate_exp<<<nodeGrid, TPB, 0, stream>>>(gate, batch, gm, gexp, gs);
        k_gate_scatter<<<nodeWGrid, TPB, 0, stream>>>(acc, batch, gexp, gs,
                                                      gout + (size_t)v * BBG * DD);
    }

    k_clf<<<BBG, DD, 0, stream>>>(gout, clf_W1, clf_b1, clf_W2, clf_b2, out);
}
